// SynCo_78194174591300
// MI455X (gfx1250) — matrix operand order repaired and checked by reading, not compiled, not run on hardware
//
#include <hip/hip_runtime.h>
#include <hip/hip_bf16.h>
#include <stdint.h>

#define DIM   128
#define BQ    512
#define KQ    65536
#define NHARD 1024
#define INV_T (1.0f / 0.07f)

typedef __bf16 bf16_t;
typedef __bf16 v8bf  __attribute__((ext_vector_type(8)));
typedef __bf16 v16bf __attribute__((ext_vector_type(16)));
typedef float  v8f   __attribute__((ext_vector_type(8)));

__device__ inline float wave_reduce_sum(float v) {
#pragma unroll
  for (int off = 16; off > 0; off >>= 1)
    v += __shfl_xor(v, off, 32);
  return v;
}

__device__ inline unsigned short f2bf(float f) {
  unsigned int u = __float_as_uint(f);
  u += 0x7FFFu + ((u >> 16) & 1u);   // round to nearest even
  return (unsigned short)(u >> 16);
}

// ---------------- K1: normalize queue rows -> bf16 Qn + inverse norms ----------------
__global__ void k_norm_queue(const float* __restrict__ queue,
                             unsigned short* __restrict__ Qbf,
                             float* __restrict__ qinv) {
  int wave = (int)((blockIdx.x * blockDim.x + threadIdx.x) >> 5);
  int lane = (int)(threadIdx.x & 31);
  if (wave >= KQ) return;
  float4 v = *(const float4*)(queue + (size_t)wave * DIM + lane * 4);
  float ss = wave_reduce_sum(v.x * v.x + v.y * v.y + v.z * v.z + v.w * v.w);
  float inv = 1.0f / fmaxf(sqrtf(ss), 1e-12f);
  unsigned short s0 = f2bf(v.x * inv), s1 = f2bf(v.y * inv);
  unsigned short s2 = f2bf(v.z * inv), s3 = f2bf(v.w * inv);
  uint2 packed;
  packed.x = (unsigned)s0 | ((unsigned)s1 << 16);
  packed.y = (unsigned)s2 | ((unsigned)s3 << 16);
  *(uint2*)(Qbf + (size_t)wave * DIM + lane * 4) = packed;
  if (lane == 0) qinv[wave] = inv;
}

// ---------------- K2: normalize q,k -> qn (f32 + bf16), l_pos ----------------
__global__ void k_norm_qk(const float* __restrict__ q, const float* __restrict__ k,
                          float* __restrict__ qn, unsigned short* __restrict__ qbf,
                          float* __restrict__ out, int ldo) {
  int wave = (int)((blockIdx.x * blockDim.x + threadIdx.x) >> 5);
  int lane = (int)(threadIdx.x & 31);
  if (wave >= BQ) return;
  float4 qv = *(const float4*)(q + (size_t)wave * DIM + lane * 4);
  float4 kv = *(const float4*)(k + (size_t)wave * DIM + lane * 4);
  float sq = wave_reduce_sum(qv.x * qv.x + qv.y * qv.y + qv.z * qv.z + qv.w * qv.w);
  float sk = wave_reduce_sum(kv.x * kv.x + kv.y * kv.y + kv.z * kv.z + kv.w * kv.w);
  float iq = 1.0f / fmaxf(sqrtf(sq), 1e-12f);
  float ik = 1.0f / fmaxf(sqrtf(sk), 1e-12f);
  float4 qnv; qnv.x = qv.x * iq; qnv.y = qv.y * iq; qnv.z = qv.z * iq; qnv.w = qv.w * iq;
  *(float4*)(qn + (size_t)wave * DIM + lane * 4) = qnv;
  unsigned short s0 = f2bf(qnv.x), s1 = f2bf(qnv.y), s2 = f2bf(qnv.z), s3 = f2bf(qnv.w);
  uint2 packed;
  packed.x = (unsigned)s0 | ((unsigned)s1 << 16);
  packed.y = (unsigned)s2 | ((unsigned)s3 << 16);
  *(uint2*)(qbf + (size_t)wave * DIM + lane * 4) = packed;
  float lp = wave_reduce_sum(qnv.x * kv.x + qnv.y * kv.y + qnv.z * kv.z + qnv.w * kv.w) * ik;
  if (lane == 0) out[(size_t)wave * ldo] = lp * INV_T;
}

// ---------------- K3: l_neg GEMM via v_wmma_f32_16x16x32_bf16 ----------------
// Each wave computes a 16(M) x 64(N) tile of out[m][1+n] = (qn . Qn) / T.
__global__ void k_gemm_wmma(const unsigned short* __restrict__ Abf,   // [512][128] bf16
                            const unsigned short* __restrict__ Bbf,   // [65536][128] bf16
                            float* __restrict__ out, int ldo) {
  const int NT = KQ / 64;  // 1024 N-tiles
  int wave = (int)(blockIdx.x * (blockDim.x >> 5) + (threadIdx.x >> 5));
  int lane = (int)(threadIdx.x & 31);
  int mt = wave / NT, nt = wave % NT;
  if (mt >= BQ / 16) return;
  int m0 = mt * 16, n0 = nt * 64;
  int lhi = lane >> 4, llo = lane & 15;

  const bf16_t* Ar = (const bf16_t*)Abf + (size_t)(m0 + llo) * DIM + 16 * lhi;
  v8f acc[4] = {};

#pragma unroll
  for (int kk = 0; kk < DIM; kk += 32) {
    v8bf alo = *(const v8bf*)(Ar + kk);
    v8bf ahi = *(const v8bf*)(Ar + kk + 8);
    v16bf a = __builtin_shufflevector(alo, ahi, 0, 1, 2, 3, 4, 5, 6, 7,
                                      8, 9, 10, 11, 12, 13, 14, 15);
#pragma unroll
    for (int s = 0; s < 4; ++s) {
      const bf16_t* Br = (const bf16_t*)Bbf + (size_t)(n0 + s * 16 + llo) * DIM + 16 * lhi + kk;
      v16bf bmat = *(const v16bf*)Br;
      acc[s] = __builtin_amdgcn_wmma_f32_16x16x32_bf16(
          false, a, false, bmat, (short)0, acc[s], false, false);
    }
  }
#pragma unroll
  for (int s = 0; s < 4; ++s) {
#pragma unroll
    for (int r = 0; r < 8; ++r) {
      int row = m0 + r + 8 * lhi;
      int col = 1 + n0 + s * 16 + llo;
      out[(size_t)row * ldo + col] = acc[s][r] * INV_T;
    }
  }
}

// ---------------- K4: per-row top-1024 radix select in LDS ----------------
__global__ void __launch_bounds__(1024) k_topk(const float* __restrict__ out, int ldo,
                                               int* __restrict__ idxs) {
  extern __shared__ unsigned int smem[];
  unsigned int* keys = smem;             // 65536
  unsigned int* hist = smem + KQ;        // 256
  unsigned int* ctrl = hist + 256;       // [0]=prefix [1]=need [2]=cntGT [3]=cntEQ
  int b = (int)blockIdx.x;
  int tid = (int)threadIdx.x;
  const float* row = out + (size_t)b * ldo + 1;

  for (int i = tid; i < KQ; i += (int)blockDim.x) {
    unsigned int u = __float_as_uint(row[i]);
    u = (u & 0x80000000u) ? ~u : (u | 0x80000000u);   // order-preserving key
    keys[i] = u;
  }
  if (tid == 0) { ctrl[0] = 0u; ctrl[1] = NHARD; ctrl[2] = 0u; ctrl[3] = 0u; }
  __syncthreads();

  for (int pass = 0; pass < 4; ++pass) {
    int shift = 24 - 8 * pass;
    for (int i = tid; i < 256; i += (int)blockDim.x) hist[i] = 0u;
    __syncthreads();
    unsigned int pfx = ctrl[0];
    unsigned int himask = (pass == 0) ? 0u : (0xFFFFFFFFu << (shift + 8));
    for (int i = tid; i < KQ; i += (int)blockDim.x) {
      unsigned int kk = keys[i];
      if ((kk & himask) == (pfx & himask))
        atomicAdd(&hist[(kk >> shift) & 255u], 1u);
    }
    __syncthreads();
    if (tid == 0) {
      unsigned int need = ctrl[1];
      unsigned int cum = 0; int bsel = 0;
      for (int bb = 255; bb >= 0; --bb) {
        unsigned int c = hist[bb];
        if (cum + c >= need) { bsel = bb; break; }
        cum += c;
      }
      ctrl[1] = need - cum;                        // rank within selected bin
      ctrl[0] = ctrl[0] | ((unsigned int)bsel << shift);
    }
    __syncthreads();
  }

  unsigned int thr = ctrl[0];
  unsigned int needEq = ctrl[1];
  for (int i = tid; i < KQ; i += (int)blockDim.x) {
    if (keys[i] > thr) {
      unsigned int pos = atomicAdd(&ctrl[2], 1u);
      idxs[(size_t)b * NHARD + pos] = i;
    }
  }
  __syncthreads();
  unsigned int ngt = ctrl[2];
  for (int i = tid; i < KQ; i += (int)blockDim.x) {
    if (keys[i] == thr) {
      unsigned int pos = atomicAdd(&ctrl[3], 1u);
      if (pos < needEq) idxs[(size_t)b * NHARD + ngt + pos] = i;
    }
  }
}

// ---------------- K5: synthesize hard negatives + l_hard ----------------
__global__ void k_synth(const float* __restrict__ qn,
                        const float* __restrict__ queue,
                        const float* __restrict__ qinv,
                        const int* __restrict__ idxs,
                        const int* __restrict__ idx1, const float* __restrict__ alpha_u,
                        const int* __restrict__ idx2, const float* __restrict__ beta_u,
                        const int* __restrict__ idx3a, const int* __restrict__ idx3b,
                        const float* __restrict__ gamma_u,
                        const int* __restrict__ idx4, const float* __restrict__ noise4,
                        const int* __restrict__ idx5, const float* __restrict__ pert5,
                        const int* __restrict__ idx6,
                        float* __restrict__ out, int ldo) {
  int b = (int)blockIdx.x;
  int wv = (int)(threadIdx.x >> 5);
  int lane = (int)(threadIdx.x & 31);
  int n = (int)blockIdx.y * 8 + wv;
  if (n >= 960) return;
  const int* hardRow = idxs + (size_t)b * NHARD;

  float qv[4];
  { float4 t = *(const float4*)(qn + (size_t)b * DIM + lane * 4);
    qv[0] = t.x; qv[1] = t.y; qv[2] = t.z; qv[3] = t.w; }

  auto gather = [&](int code, float* g) {
    int r = hardRow[code];
    float4 t = *(const float4*)(queue + (size_t)r * DIM + lane * 4);
    float s = qinv[r];
    g[0] = t.x * s; g[1] = t.y * s; g[2] = t.z * s; g[3] = t.w * s;
  };

  float h[4];
  if (n < 256) {                       // type 1: interpolate toward q
    int j = n;
    float a = alpha_u[(size_t)b * 256 + j] * 0.5f;
    float g[4]; gather(idx1[(size_t)b * 256 + j], g);
#pragma unroll
    for (int i = 0; i < 4; ++i) h[i] = a * qv[i] + (1.0f - a) * g[i];
  } else if (n < 512) {                // type 2: extrapolate away from q
    int j = n - 256;
    float bb = 1.0f + beta_u[(size_t)b * 256 + j] * 0.5f;
    float g[4]; gather(idx2[(size_t)b * 256 + j], g);
#pragma unroll
    for (int i = 0; i < 4; ++i) h[i] = qv[i] + bb * (g[i] - qv[i]);
  } else if (n < 768) {                // type 3: mixup of two hard negatives
    int j = n - 512;
    float gm = gamma_u[(size_t)b * 256 + j];
    float ga[4], gb[4];
    gather(idx3a[(size_t)b * 256 + j], ga);
    gather(idx3b[(size_t)b * 256 + j], gb);
#pragma unroll
    for (int i = 0; i < 4; ++i) h[i] = gm * ga[i] + (1.0f - gm) * gb[i];
  } else if (n < 832) {                // type 4: noise injection
    int j = n - 768;
    float g[4]; gather(idx4[(size_t)b * 64 + j], g);
    float4 nz = *(const float4*)(noise4 + ((size_t)b * 64 + j) * DIM + lane * 4);
    h[0] = g[0] + 0.1f * nz.x; h[1] = g[1] + 0.1f * nz.y;
    h[2] = g[2] + 0.1f * nz.z; h[3] = g[3] + 0.1f * nz.w;
  } else if (n < 896) {                // type 5: finite-difference gradient perturbation
    int j = n - 832;
    float g[4]; gather(idx5[(size_t)b * 64 + j], g);
    float4 pv = *(const float4*)(pert5 + ((size_t)b * 64 + j) * DIM + lane * 4);
    float p[4] = { pv.x, pv.y, pv.z, pv.w };
    float ag = wave_reduce_sum(qv[0] * p[0] + qv[1] * p[1] + qv[2] * p[2] + qv[3] * p[3]);
    float cf = 0.01f * ag * 1e-5f;     // DELTA * approx_grad * EPS
#pragma unroll
    for (int i = 0; i < 4; ++i) h[i] = g[i] + cf * p[i];
  } else {                             // type 6: adversarial step along q
    int j = n - 896;
    float g[4]; gather(idx6[(size_t)b * 64 + j], g);
#pragma unroll
    for (int i = 0; i < 4; ++i) h[i] = g[i] + 0.01f * qv[i];
  }

  float ss = wave_reduce_sum(h[0] * h[0] + h[1] * h[1] + h[2] * h[2] + h[3] * h[3]);
  float dq = wave_reduce_sum(h[0] * qv[0] + h[1] * qv[1] + h[2] * qv[2] + h[3] * qv[3]);
  if (lane == 0)
    out[(size_t)b * ldo + 1 + KQ + n] = dq / fmaxf(sqrtf(ss), 1e-12f) * INV_T;
}

extern "C" void kernel_launch(void* const* d_in, const int* in_sizes, int n_in,
                              void* d_out, int out_size, void* d_ws, size_t ws_size,
                              hipStream_t stream) {
  const float* q       = (const float*)d_in[0];
  const float* k       = (const float*)d_in[1];
  const float* queue   = (const float*)d_in[2];
  const int*   idx1    = (const int*)d_in[3];
  const float* alpha_u = (const float*)d_in[4];
  const int*   idx2    = (const int*)d_in[5];
  const float* beta_u  = (const float*)d_in[6];
  const int*   idx3a   = (const int*)d_in[7];
  const int*   idx3b   = (const int*)d_in[8];
  const float* gamma_u = (const float*)d_in[9];
  const int*   idx4    = (const int*)d_in[10];
  const float* noise4  = (const float*)d_in[11];
  const int*   idx5    = (const int*)d_in[12];
  const float* pert5   = (const float*)d_in[13];
  const int*   idx6    = (const int*)d_in[14];
  float* out = (float*)d_out;
  int ldo = out_size / BQ;             // 1 + 65536 + 960

  char* ws = (char*)d_ws;
  size_t off = 0;
  unsigned short* Qbf = (unsigned short*)(ws + off); off += (size_t)KQ * DIM * 2;
  unsigned short* qbf = (unsigned short*)(ws + off); off += (size_t)BQ * DIM * 2;
  float* qn   = (float*)(ws + off); off += (size_t)BQ * DIM * 4;
  float* qinv = (float*)(ws + off); off += (size_t)KQ * 4;
  int*   idxs = (int*)(ws + off);   off += (size_t)BQ * NHARD * 4;

  k_norm_queue<<<dim3(KQ / 8), dim3(256), 0, stream>>>(queue, Qbf, qinv);
  k_norm_qk<<<dim3(BQ / 8), dim3(256), 0, stream>>>(q, k, qn, qbf, out, ldo);
  k_gemm_wmma<<<dim3((BQ / 16) * (KQ / 64) / 8), dim3(256), 0, stream>>>(qbf, Qbf, out, ldo);
  size_t smem = (size_t)(KQ + 256 + 8) * sizeof(unsigned int);
  k_topk<<<dim3(BQ), dim3(1024), smem, stream>>>(out, ldo, idxs);
  k_synth<<<dim3(BQ, 120), dim3(256), 0, stream>>>(qn, queue, qinv, idxs,
      idx1, alpha_u, idx2, beta_u, idx3a, idx3b, gamma_u,
      idx4, noise4, idx5, pert5, idx6, out, ldo);
}
